// Transformer_54571854463432
// MI455X (gfx1250) — matrix operand order repaired and checked by reading, not compiled, not run on hardware
//
#include <hip/hip_runtime.h>
#include <hip/hip_bf16.h>

// ---------------------------------------------------------------------------
// Problem constants
#define BATCH   128
#define CHAN    256
#define HW      784           // 28*28
#define HWPAD   800           // K padded to multiple of 32 for WMMA
#define LDSK    808           // LDS row stride in bf16 elems (bank-conflict free)
#define NIMG    (BATCH*CHAN)  // 32768 images of 784
#define NTOT    ((size_t)NIMG*HW)
#define NLAYER  3
#define WELEM   ((size_t)HW*HWPAD)   // one padded bf16 weight matrix: 784*800
#define NTILE   49            // N tiles of 16 (784/16)
#define TPW     6             // unconditional N-tiles per wave (8*6=48; tile 48 is the tail)

typedef __attribute__((ext_vector_type(16))) __bf16 v16bf;
typedef __attribute__((ext_vector_type(8)))  float  v8f;

struct __align__(16) U4 { unsigned x, y, z, w; };

// ---- WMMA fragment loaders (wave32, 16x16x32 bf16) ------------------------
// A (16x32, MxK), 16-bit layout: lanes 0-15 -> M=lane, K {kb..kb+7, kb+16..kb+23}
//                                lanes 16-31 -> M=lane-16, K {kb+8..kb+15, kb+24..kb+31}
static __device__ __forceinline__ v16bf load_a_frag(const __bf16* base, int ldk,
                                                    int kb, int lane) {
    int half = lane >> 4;
    int mrow = lane & 15;
    const __bf16* rp = base + mrow * ldk;
    union { v16bf v; U4 u[2]; } f;
    f.u[0] = *(const U4*)(rp + kb + half * 8);
    f.u[1] = *(const U4*)(rp + kb + 16 + half * 8);
    return f.v;
}

// B (32x16, KxN) with B[k][n] = W[n0+n][k], W row-major (N, ldk) bf16:
// lanes 0-15 -> N=lane, K {kb..kb+7, kb+16..kb+23} ; lanes 16-31 -> N=lane-16, K {kb+8..kb+15, kb+24..kb+31}
static __device__ __forceinline__ v16bf load_b_frag(const __bf16* W, int ldk,
                                                    int n0, int kb, int lane) {
    int half = lane >> 4;
    int nn   = lane & 15;
    const __bf16* p = W + (size_t)(n0 + nn) * ldk + kb + half * 8;
    union { v16bf v; U4 u[2]; } f;
    f.u[0] = *(const U4*)(p);
    f.u[1] = *(const U4*)(p + 16);
    return f.v;
}

// ---------------------------------------------------------------------------
// 1) xbuf = x + sin(position)
__global__ void prep_pe_kernel(const float* __restrict__ x,
                               float* __restrict__ xbuf, size_t n) {
    size_t i = (size_t)blockIdx.x * blockDim.x + threadIdx.x;
    if (i < n) {
        int p = (int)(i % HW);
        xbuf[i] = x[i] + sinf((float)p);
    }
}

// 2) Convert W1s/W2s (NLAYER x 784x784 f32) -> bf16 padded (N=784, K=800)
__global__ void prep_wbf_kernel(const float* __restrict__ W1s,
                                const float* __restrict__ W2s,
                                __bf16* __restrict__ Wbf) {
    size_t i = (size_t)blockIdx.x * blockDim.x + threadIdx.x;
    size_t tot = 6 * WELEM;
    if (i >= tot) return;
    int m   = (int)(i / WELEM);        // 0..5 : 3x W1 then 3x W2
    size_t r = i % WELEM;
    int n = (int)(r / HWPAD);
    int k = (int)(r % HWPAD);
    float v = 0.0f;
    if (k < HW) {
        const float* src = (m < 3) ? (W1s + (size_t)m * HW * HW)
                                   : (W2s + (size_t)(m - 3) * HW * HW);
        v = src[(size_t)n * HW + k];
    }
    Wbf[i] = (__bf16)v;
}

// 3) Attention prep: only rows (b=l, c=0) of qkv are ever used.
//    One block per layer l. Produces av[l][p][i][m] (p = r*4+c), 784 floats each.
__global__ void attn_prep_kernel(const float* __restrict__ x,
                                 const float* __restrict__ Wqkv,
                                 const float* __restrict__ bqkv,
                                 float* __restrict__ av) {
    __shared__ float xrow[HW];
    __shared__ float qkv[2352];
    int l   = blockIdx.x;
    int tid = threadIdx.x;
    const float* xr = x + (size_t)l * CHAN * HW;   // image (b=l, c=0)
    for (int p = tid; p < HW; p += blockDim.x) xrow[p] = xr[p];
    __syncthreads();
    for (int o = tid; o < 2352; o += blockDim.x) {
        const float* wr = Wqkv + (size_t)o * HW;
        float d = bqkv[o];
        for (int k = 0; k < HW; ++k) d += wr[k] * xrow[k];
        qkv[o] = d;
    }
    __syncthreads();
    // qkv reshaped (16, 3, 7, 7): [p][t][i][j] = qkv[p*147 + t*49 + i*7 + j]
    if (tid < 112) {
        int p = tid / 7, i = tid % 7;
        const float* qb = qkv + p * 147;
        float qrow[7];
#pragma unroll
        for (int j = 0; j < 7; ++j) qrow[j] = qb[i * 7 + j];
        float sc[7], mx = -3.4e38f;
#pragma unroll
        for (int kk = 0; kk < 7; ++kk) {
            float d = 0.f;
#pragma unroll
            for (int j = 0; j < 7; ++j) d += qrow[j] * qb[49 + kk * 7 + j];
            d *= 0.25f;                 // P^-0.5 = 16^-0.5
            sc[kk] = d;
            mx = fmaxf(mx, d);
        }
        float s = 0.f;
#pragma unroll
        for (int kk = 0; kk < 7; ++kk) { sc[kk] = __expf(sc[kk] - mx); s += sc[kk]; }
        float inv = 1.0f / s;
#pragma unroll
        for (int m = 0; m < 7; ++m) {
            float a = 0.f;
#pragma unroll
            for (int kk = 0; kk < 7; ++kk) a += sc[kk] * qb[98 + kk * 7 + m];
            av[(size_t)l * HW + p * 49 + i * 7 + m] = a * inv;
        }
    }
}

// 4) Per-image: LN -> per-tile 7x7 multiply with av, write pre-softmax h in the
//    FINAL scrambled layout: out[b,ch, c*7+i, r*7+m] with p = r*4+c.
__global__ void attn_apply_kernel(const float* __restrict__ xbuf,
                                  const float* __restrict__ gamma,
                                  const float* __restrict__ beta,
                                  const float* __restrict__ avl,
                                  float* __restrict__ hpre) {
    __shared__ float ximg[HW];
    __shared__ float avs[HW];
    __shared__ float r1[256], r2[256];
    int tid = threadIdx.x;
    size_t base = (size_t)blockIdx.x * HW;
    float s1 = 0.f, s2 = 0.f;
    for (int p = tid; p < HW; p += 256) {
        float v = xbuf[base + p];
        ximg[p] = v; s1 += v; s2 += v * v;
    }
    r1[tid] = s1; r2[tid] = s2;
    __syncthreads();
    for (int off = 128; off > 0; off >>= 1) {
        if (tid < off) { r1[tid] += r1[tid + off]; r2[tid] += r2[tid + off]; }
        __syncthreads();
    }
    float mean = r1[0] * (1.0f / HW);
    float var  = r2[0] * (1.0f / HW) - mean * mean;
    float inv  = rsqrtf(var + 1e-5f);
    for (int p = tid; p < HW; p += 256)
        ximg[p] = (ximg[p] - mean) * inv * gamma[p] + beta[p];
    for (int p = tid; p < HW; p += 256) avs[p] = avl[p];
    __syncthreads();
    for (int p = tid; p < HW; p += 256) {
        int orow = p / 28, ocol = p % 28;
        int c = orow / 7, i = orow % 7;
        int r = ocol / 7, m = ocol % 7;
        const float* x1t = ximg + (r * 7 + i) * 28 + c * 7;   // x1[r*7+i][c*7+j]
        const float* at  = avs + (r * 4 + c) * 49 + m;        // av[p][j][m]
        float h = 0.f;
#pragma unroll
        for (int j = 0; j < 7; ++j) h += x1t[j] * at[j * 7];
        hpre[base + p] = h;
    }
}

// 5) Softmax over channel dim (256) per (b, spatial pos) + residual into xbuf.
__global__ void chan_softmax_kernel(const float* __restrict__ hpre,
                                    float* __restrict__ xbuf) {
    int b = blockIdx.x >> 2;
    int p = (blockIdx.x & 3) * 256 + threadIdx.x;
    if (p >= HW) return;
    const float* hb = hpre + (size_t)b * CHAN * HW + p;
    float m = -3.4e38f, s = 0.f;
    for (int ch = 0; ch < CHAN; ++ch) {
        float v  = hb[(size_t)ch * HW];
        float nm = fmaxf(m, v);
        s = s * __expf(m - nm) + __expf(v - nm);
        m = nm;
    }
    float invs = 1.0f / s;
    float* xb = xbuf + (size_t)b * CHAN * HW + p;
    for (int ch = 0; ch < CHAN; ++ch) {
        float v = hb[(size_t)ch * HW];
        xb[(size_t)ch * HW] += __expf(v - m) * invs;
    }
}

// 6) Fused MLP: LN -> GEMM1(bf16 WMMA)+leaky_relu -> GEMM2(bf16 WMMA)+residual.
//    One block = 16 images (rows). 256 threads = 8 waves.
//    Main K loop: 6 unconditional WMMAs per wave per K-step (branch-free body);
//    the 49th N-tile (nt=48) runs as a small tail loop on wave 0 only.
__global__ void __launch_bounds__(256)
mlp_fused_kernel(float* __restrict__ xbuf,
                 const __bf16* __restrict__ W1,
                 const __bf16* __restrict__ W2,
                 const float* __restrict__ b1,
                 const float* __restrict__ b2,
                 const float* __restrict__ gamma,
                 const float* __restrict__ beta) {
    __shared__ __align__(16) __bf16 sA[16 * LDSK];   // x3 (post-LN) bf16
    __shared__ __align__(16) __bf16 sH[16 * LDSK];   // hidden bf16
    int tid  = threadIdx.x;
    int lane = tid & 31;
    int w    = tid >> 5;                 // wave id 0..7
    int half = lane >> 4;
    size_t rowbase = (size_t)blockIdx.x * 16;

    // ---- Phase A: layernorm 16 rows, store bf16 into sA (K padded w/ zeros)
    for (int rr = 0; rr < 2; ++rr) {
        int row = 2 * w + rr;
        const float* xr = xbuf + (rowbase + row) * HW;
        float s1 = 0.f, s2 = 0.f;
        for (int p = lane; p < HW; p += 32) {
            float v = xr[p];
            s1 += v; s2 += v * v;
        }
#pragma unroll
        for (int off = 16; off > 0; off >>= 1) {
            s1 += __shfl_xor(s1, off, 32);
            s2 += __shfl_xor(s2, off, 32);
        }
        float mean = s1 * (1.0f / HW);
        float var  = s2 * (1.0f / HW) - mean * mean;
        float inv  = rsqrtf(var + 1e-5f);
        for (int p = lane; p < HWPAD; p += 32) {
            float v = 0.f;
            if (p < HW) v = (xr[p] - mean) * inv * gamma[p] + beta[p];
            sA[row * LDSK + p] = (__bf16)v;
        }
    }
    // zero the K-pad region of sH (cols 784..799 of all 16 rows)
    if (tid < 256) sH[(tid >> 4) * LDSK + HW + (tid & 15)] = (__bf16)0.0f;
    __syncthreads();

    // ---- Phase B: hid = leaky_relu(x3 @ W1^T + b1)  -> sH (bf16)
    {
        v8f acc[TPW] = {};
#pragma unroll 1
        for (int kb = 0; kb < HWPAD; kb += 32) {
            v16bf a = load_a_frag(sA, LDSK, kb, lane);
#pragma unroll
            for (int t = 0; t < TPW; ++t) {      // nt = t*8+w <= 47 < 49 always
                v16bf b = load_b_frag(W1, HWPAD, (t * 8 + w) * 16, kb, lane);
                acc[t] = __builtin_amdgcn_wmma_f32_16x16x32_bf16(
                    false, a, false, b, (short)0, acc[t], false, false);
            }
        }
#pragma unroll
        for (int t = 0; t < TPW; ++t) {
            int n = (t * 8 + w) * 16 + (lane & 15);
            float bias = b1[n];
#pragma unroll
            for (int v = 0; v < 8; ++v) {
                float val = acc[t][v] + bias;
                val = (val >= 0.f) ? val : 0.01f * val;   // leaky_relu
                sH[(v + 8 * half) * LDSK + n] = (__bf16)val;
            }
        }
        if (w == 0) {                            // tail N-tile nt=48
            v8f acc2 = {};
#pragma unroll 1
            for (int kb = 0; kb < HWPAD; kb += 32) {
                v16bf a = load_a_frag(sA, LDSK, kb, lane);
                v16bf b = load_b_frag(W1, HWPAD, 48 * 16, kb, lane);
                acc2 = __builtin_amdgcn_wmma_f32_16x16x32_bf16(
                    false, a, false, b, (short)0, acc2, false, false);
            }
            int n = 48 * 16 + (lane & 15);
            float bias = b1[n];
#pragma unroll
            for (int v = 0; v < 8; ++v) {
                float val = acc2[v] + bias;
                val = (val >= 0.f) ? val : 0.01f * val;
                sH[(v + 8 * half) * LDSK + n] = (__bf16)val;
            }
        }
    }
    __syncthreads();

    // ---- Phase C: x4 = hid @ W2^T + b2 ; xbuf = x2 + x4 (in place, rows owned)
    {
        v8f acc[TPW] = {};
#pragma unroll 1
        for (int kb = 0; kb < HWPAD; kb += 32) {
            v16bf a = load_a_frag(sH, LDSK, kb, lane);
#pragma unroll
            for (int t = 0; t < TPW; ++t) {
                v16bf b = load_b_frag(W2, HWPAD, (t * 8 + w) * 16, kb, lane);
                acc[t] = __builtin_amdgcn_wmma_f32_16x16x32_bf16(
                    false, a, false, b, (short)0, acc[t], false, false);
            }
        }
#pragma unroll
        for (int t = 0; t < TPW; ++t) {
            int n = (t * 8 + w) * 16 + (lane & 15);
            float bias = b2[n];
#pragma unroll
            for (int v = 0; v < 8; ++v) {
                size_t idx = (rowbase + v + 8 * half) * HW + n;
                xbuf[idx] = xbuf[idx] + acc[t][v] + bias;
            }
        }
        if (w == 0) {                            // tail N-tile nt=48
            v8f acc2 = {};
#pragma unroll 1
            for (int kb = 0; kb < HWPAD; kb += 32) {
                v16bf a = load_a_frag(sH, LDSK, kb, lane);
                v16bf b = load_b_frag(W2, HWPAD, 48 * 16, kb, lane);
                acc2 = __builtin_amdgcn_wmma_f32_16x16x32_bf16(
                    false, a, false, b, (short)0, acc2, false, false);
            }
            int n = 48 * 16 + (lane & 15);
            float bias = b2[n];
#pragma unroll
            for (int v = 0; v < 8; ++v) {
                size_t idx = (rowbase + v + 8 * half) * HW + n;
                xbuf[idx] = xbuf[idx] + acc2[v] + bias;
            }
        }
    }
}

// 7) out = tanh(xbuf)
__global__ void final_tanh_kernel(const float* __restrict__ xbuf,
                                  float* __restrict__ out, size_t n) {
    size_t i = (size_t)blockIdx.x * blockDim.x + threadIdx.x;
    if (i < n) out[i] = tanhf(xbuf[i]);
}

// ---------------------------------------------------------------------------
extern "C" void kernel_launch(void* const* d_in, const int* in_sizes, int n_in,
                              void* d_out, int out_size, void* d_ws, size_t ws_size,
                              hipStream_t stream) {
    const float* x     = (const float*)d_in[0];
    const float* Wqkv  = (const float*)d_in[1];
    const float* bqkv  = (const float*)d_in[2];
    const float* gamma = (const float*)d_in[3];
    const float* beta  = (const float*)d_in[4];
    const float* W1s   = (const float*)d_in[5];
    const float* b1s   = (const float*)d_in[6];
    const float* W2s   = (const float*)d_in[7];
    const float* b2s   = (const float*)d_in[8];
    float* out = (float*)d_out;

    // workspace layout (16B-aligned sections)
    const size_t XBUF_BYTES = NTOT * sizeof(float);         // 102.76 MB
    const size_t WBF_BYTES  = 6 * WELEM * sizeof(__bf16);   // 7.53 MB
    char* ws    = (char*)d_ws;
    float*  xbuf = (float*)ws;
    __bf16* Wbf  = (__bf16*)(ws + XBUF_BYTES);
    float*  av   = (float*)(ws + XBUF_BYTES + WBF_BYTES);
    float*  hpre = out;   // d_out doubles as pre-softmax scratch until the end

    const int T = 256;
    prep_pe_kernel<<<(unsigned)((NTOT + T - 1) / T), T, 0, stream>>>(x, xbuf, NTOT);
    {
        size_t n = 6 * WELEM;
        prep_wbf_kernel<<<(unsigned)((n + T - 1) / T), T, 0, stream>>>(W1s, W2s, Wbf);
    }
    attn_prep_kernel<<<NLAYER, T, 0, stream>>>(x, Wqkv, bqkv, av);

    for (int l = 0; l < NLAYER; ++l) {
        attn_apply_kernel<<<NIMG, T, 0, stream>>>(xbuf, gamma, beta,
                                                  av + (size_t)l * HW, hpre);
        chan_softmax_kernel<<<BATCH * 4, T, 0, stream>>>(hpre, xbuf);
        mlp_fused_kernel<<<NIMG / 16, T, 0, stream>>>(
            xbuf,
            Wbf + (size_t)l * WELEM,
            Wbf + (size_t)(3 + l) * WELEM,
            b1s + (size_t)l * HW,
            b2s + (size_t)l * HW,
            gamma, beta);
    }
    final_tanh_kernel<<<(unsigned)((NTOT + T - 1) / T), T, 0, stream>>>(xbuf, out, NTOT);
}
